// RNNLanguage_35003983462991
// MI455X (gfx1250) — compile-verified, matrix operand order checked by reading, not run on hardware
//
#include <hip/hip_runtime.h>
#include <hip/hip_bf16.h>

// Sizes from the reference
#define Hh   512
#define Bb   32
#define Tt   512
#define Vv   50257
#define NWG  16          // workgroups in the scan cluster
#define NT4  3142        // ceil(V/16) n-tiles for logits

typedef __attribute__((ext_vector_type(16))) __bf16 v16bf;
typedef __attribute__((ext_vector_type(8)))  float  v8f;

__device__ inline __bf16 f2bf(float f) {
  unsigned u = __builtin_bit_cast(unsigned, f);
  unsigned r = u + 0x7FFFu + ((u >> 16) & 1u);      // round-to-nearest-even
  unsigned short h = (unsigned short)(r >> 16);
  return __builtin_bit_cast(__bf16, h);
}

// act(x) = copysign(1 - 1/(|x|^2 + |x| + 1), x)
__device__ inline float act_f(float x) {
  float xa = fabsf(x);
  float d  = fmaf(xa, xa, xa) + 1.0f;
  float v  = 1.0f - 1.0f / d;
  return copysignf(v, x);
}

// 16-bit WMMA A/B fragment K index for vector element e, lane half hi (ISA 7.12.2)
__device__ inline int kmap(int e, int hi) {
  return ((e & 8) << 1) + hi * 8 + (e & 7);
}

__device__ inline v8f wmma_bf16(v16bf a, v16bf b, v8f c) {
  return __builtin_amdgcn_wmma_f32_16x16x32_bf16(false, a, false, b, (short)0, c,
                                                 false, false);
}

// ---------------------------------------------------------------------------
// K0: row norms of hh -> s[n] = 1/(||hh[n,:]|| * H); zero stateF; init barrier
// grid 512 x 128
__global__ __launch_bounds__(128) void k_prep(const float* __restrict__ hh,
                                              float* __restrict__ s,
                                              __bf16* __restrict__ stateF,
                                              unsigned* __restrict__ bar) {
  int n = blockIdx.x;
  __shared__ float red[128];
  float acc = 0.f;
  for (int j = threadIdx.x; j < Hh; j += 128) {
    float v = hh[n * Hh + j];
    acc += v * v;
  }
  red[threadIdx.x] = acc;
  __syncthreads();
  for (int off = 64; off > 0; off >>= 1) {
    if (threadIdx.x < off) red[threadIdx.x] += red[threadIdx.x + off];
    __syncthreads();
  }
  if (threadIdx.x == 0) s[n] = 1.0f / (sqrtf(red[0]) * (float)Hh);
  // zero initial RNN state (fragment layout, 16384 bf16 total = 512 blocks * 32)
  if (threadIdx.x < 32) stateF[n * 32 + threadIdx.x] = f2bf(0.f);
  if (n == 0 && threadIdx.x < 2) bar[threadIdx.x] = 0u;
}

// ---------------------------------------------------------------------------
// K1: pack ih and scaledHH into WMMA-B fragment layout (bf16):
//     frag addr = ((nt*16 + kt)*32 + lane)*16 + e ; B[k][n], n = nt*16+lane&15
// grid 1024 x 256
__global__ __launch_bounds__(256) void k_pack(const float* __restrict__ ih,
                                              const float* __restrict__ hh,
                                              const float* __restrict__ s,
                                              __bf16* __restrict__ ihF,
                                              __bf16* __restrict__ hhF) {
  int idx = blockIdx.x * 256 + threadIdx.x;
  if (idx >= Hh * Hh) return;
  int e = idx & 15, lane = (idx >> 4) & 31, kt = (idx >> 9) & 15, nt = idx >> 13;
  int n = nt * 16 + (lane & 15);
  int k = kt * 32 + kmap(e, lane >> 4);
  ihF[idx] = f2bf(ih[k * Hh + n]);
  hhF[idx] = f2bf(hh[k * Hh + n] * s[n]);   // column n scaled by 1/(||hh[n,:]||*H)
}

// ---------------------------------------------------------------------------
// K2: xs[t*B+b][:] = act(E[tokens[b][t]]) @ ih + bias  (only gathered rows!)
// M = B*T = 16384, one 16-row M-tile per block; 4 waves x 8 n-tiles each.
// grid 1024 x 128
__global__ __launch_bounds__(128) void k_embed(const int* __restrict__ tokens,
                                               const float* __restrict__ emb,
                                               const __bf16* __restrict__ ihF,
                                               const float* __restrict__ bias,
                                               float* __restrict__ xs) {
  __shared__ __bf16 As[16][Hh];          // 16 KB: act(E[row]) in bf16
  int m0 = blockIdx.x * 16;
  for (int i = threadIdx.x; i < 16 * Hh; i += 128) {
    int row = i >> 9, col = i & (Hh - 1);
    int r = m0 + row, t = r >> 5, b = r & 31;
    int tok = tokens[b * Tt + t];
    As[row][col] = f2bf(act_f(emb[tok * Hh + col]));
  }
  __syncthreads();

  int wave = threadIdx.x >> 5, lane = threadIdx.x & 31;
  int mrow = lane & 15, hi = lane >> 4;
  for (int i = 0; i < 8; i++) {
    int nt = wave * 8 + i, n0 = nt * 16;
    float bv = bias[n0 + (lane & 15)];
    v8f acc;
    #pragma unroll
    for (int g = 0; g < 8; g++) acc[g] = bv;   // C[m][n] = bias[n]
    for (int kt = 0; kt < 16; kt++) {
      v16bf a;
      #pragma unroll
      for (int e = 0; e < 16; e++) a[e] = As[mrow][kt * 32 + kmap(e, hi)];
      v16bf bf = *(const v16bf*)(ihF + ((nt * 16 + kt) * 32 + lane) * 16);
      acc = wmma_bf16(a, bf, acc);
    }
    #pragma unroll
    for (int g = 0; g < 8; g++) {
      int r = m0 + hi * 8 + g;                 // D row = hi*8+g
      xs[r * Hh + n0 + (lane & 15)] = acc[g];
    }
  }
}

// ---------------------------------------------------------------------------
// Chip-wide per-step sync for the scan:
//  - launched as a 16-WG cluster: hardware cluster barrier (s_barrier_* -3)
//  - plain launch (ClusterID==0): L2-atomic split barrier fallback
__device__ inline void grid_sync(unsigned* bar) {
  __threadfence();
  __syncthreads();
  if (__builtin_amdgcn_cluster_id_x() != 0) {
    if (threadIdx.x < 32) __builtin_amdgcn_s_cluster_barrier();  // wave 0 signals+waits
    __syncthreads();
  } else {
    if (threadIdx.x == 0) {
      unsigned g = __hip_atomic_load(bar + 1, __ATOMIC_ACQUIRE, __HIP_MEMORY_SCOPE_AGENT);
      unsigned old = __hip_atomic_fetch_add(bar, 1u, __ATOMIC_ACQ_REL, __HIP_MEMORY_SCOPE_AGENT);
      if (old == (unsigned)(NWG - 1)) {
        __hip_atomic_store(bar, 0u, __ATOMIC_RELAXED, __HIP_MEMORY_SCOPE_AGENT);
        __hip_atomic_fetch_add(bar + 1, 1u, __ATOMIC_RELEASE, __HIP_MEMORY_SCOPE_AGENT);
      } else {
        while (__hip_atomic_load(bar + 1, __ATOMIC_ACQUIRE, __HIP_MEMORY_SCOPE_AGENT) == g)
          __builtin_amdgcn_s_sleep(1);
      }
      __threadfence();
    }
    __syncthreads();
  }
}

// ---------------------------------------------------------------------------
// K3: RNN scan. 16 WGs x 128 threads. WG w owns columns [w*32, w*32+32).
// state kept in global stateF in A-fragment bf16 layout; B slice in VGPRs.
// grid 16 x 128
__global__ __launch_bounds__(128, 1) void k_scan(const __bf16* __restrict__ hhF,
                                                 const float* __restrict__ xs,
                                                 __bf16* __restrict__ stateF,
                                                 unsigned* __restrict__ bar) {
  int wg = blockIdx.x;                 // 0..15
  int wave = threadIdx.x >> 5, lane = threadIdx.x & 31;
  int mt = wave & 1, ntl = wave >> 1;  // wave -> (m-tile, local n-tile)
  int nt = wg * 2 + ntl;               // global n-tile
  int n0 = nt * 16;
  int hi = lane >> 4;

  __shared__ __bf16 ns[32][33];        // new state block [b][local n], 2.1 KB

  // Preload this wave's B fragments (scaledHH column slice): 16 x v16bf = 128 VGPRs
  v16bf Bf[16];
  #pragma unroll
  for (int kt = 0; kt < 16; kt++)
    Bf[kt] = *(const v16bf*)(hhF + ((nt * 16 + kt) * 32 + lane) * 16);

  #pragma unroll 1
  for (int t = 0; t < Tt; t++) {
    v8f acc;
    #pragma unroll
    for (int g = 0; g < 8; g++) acc[g] = 0.f;
    #pragma unroll
    for (int kt = 0; kt < 16; kt++) {
      v16bf a = *(const v16bf*)(stateF + ((mt * 16 + kt) * 32 + lane) * 16);
      acc = wmma_bf16(a, Bf[kt], acc);
    }
    // new = act(e_t + state@scaledHH); stage bf16 block in LDS
    int nn = n0 + (lane & 15);
    #pragma unroll
    for (int g = 0; g < 8; g++) {
      int b = mt * 16 + hi * 8 + g;
      float v = acc[g] + xs[(t * Bb + b) * Hh + nn];
      ns[b][nn - wg * 32] = f2bf(act_f(v));
    }
    __syncthreads();
    // transpose block into fragment layout; WG w owns k-tile kt == wg
    if (threadIdx.x < 64) {
      int mtp = threadIdx.x >> 5, lanep = threadIdx.x & 31, hip2 = lanep >> 4;
      v16bf frag;
      #pragma unroll
      for (int e = 0; e < 16; e++)
        frag[e] = ns[mtp * 16 + (lanep & 15)][kmap(e, hip2)];
      *(v16bf*)(stateF + ((mtp * 16 + wg) * 32 + lanep) * 16) = frag;
    }
    grid_sync(bar);
  }
}

// ---------------------------------------------------------------------------
// K4: logits = final_state @ E^T. Streams 103 MB of embedding (HBM-bound).
// Each wave: one n-tile (16 vocab cols) x both m-tiles.  grid 786 x 128
__global__ __launch_bounds__(128) void k_logits(const __bf16* __restrict__ stateF,
                                                const float* __restrict__ emb,
                                                float* __restrict__ out) {
  int lane = threadIdx.x & 31, wave = threadIdx.x >> 5;
  int nt = blockIdx.x * 4 + wave;
  if (nt >= NT4) return;               // wave-uniform exit (EXEC stays all-1 for WMMA)
  int hi = lane >> 4;
  int n = nt * 16 + (lane & 15);
  int nc = (n < Vv) ? n : (Vv - 1);    // clamp loads on the ragged last tile

  v8f acc0, acc1;
  #pragma unroll
  for (int g = 0; g < 8; g++) { acc0[g] = 0.f; acc1[g] = 0.f; }

  for (int kt = 0; kt < 16; kt++) {
    __builtin_prefetch(emb + nc * Hh + (((kt + 1) & 15) * 32), 0, 1);
    v16bf a0 = *(const v16bf*)(stateF + ((0  + kt) * 32 + lane) * 16);
    v16bf a1 = *(const v16bf*)(stateF + ((16 + kt) * 32 + lane) * 16);
    v16bf bf;
    #pragma unroll
    for (int p = 0; p < 8; p++) {      // B[k][n] = E[n][k], consecutive k pairs
      int k = kt * 32 + kmap(2 * p, hi);
      float2 two = *(const float2*)(emb + nc * Hh + k);
      bf[2 * p]     = f2bf(two.x);
      bf[2 * p + 1] = f2bf(two.y);
    }
    acc0 = wmma_bf16(a0, bf, acc0);
    acc1 = wmma_bf16(a1, bf, acc1);
  }
  if (n < Vv) {
    #pragma unroll
    for (int g = 0; g < 8; g++) {
      out[(hi * 8 + g) * Vv + n]        = acc0[g];   // b = 0..15
      out[(16 + hi * 8 + g) * Vv + n]   = acc1[g];   // b = 16..31
    }
  }
}

// ---------------------------------------------------------------------------
extern "C" void kernel_launch(void* const* d_in, const int* in_sizes, int n_in,
                              void* d_out, int out_size, void* d_ws, size_t ws_size,
                              hipStream_t stream) {
  const int*   tokens = (const int*)  d_in[0];   // [32,512]
  const float* emb    = (const float*)d_in[1];   // [50257,512]
  const float* ih     = (const float*)d_in[2];   // [512,512]
  const float* hh     = (const float*)d_in[3];   // [512,512]
  const float* bias   = (const float*)d_in[4];   // [512]
  float* out = (float*)d_out;                    // [32,50257]

  char* ws = (char*)d_ws;
  float*    s      = (float*)   (ws);                                  //   2 KB
  __bf16*   ihF    = (__bf16*)  (ws + (4u << 10));                     // 512 KB
  __bf16*   hhF    = (__bf16*)  (ws + (4u << 10) + (512u << 10));      // 512 KB
  __bf16*   stateF = (__bf16*)  (ws + (4u << 10) + (1024u << 10));     //  32 KB
  unsigned* bar    = (unsigned*)(ws + (4u << 10) + (1056u << 10));     //   8 B
  float*    xs     = (float*)   (ws + (2u << 20));                     // 33.5 MB

  k_prep  <<<512,  128, 0, stream>>>(hh, s, stateF, bar);
  k_pack  <<<1024, 256, 0, stream>>>(ih, hh, s, ihF, hhF);
  k_embed <<<1024, 128, 0, stream>>>(tokens, emb, ihF, bias, xs);
  k_scan  <<<NWG,  128, 0, stream>>>(hhF, xs, stateF, bar);
  k_logits<<<(NT4 + 3) / 4, 128, 0, stream>>>(stateF, emb, out);
}
